// MoE_5265629905213
// MI455X (gfx1250) — matrix operand order repaired and checked by reading, not compiled, not run on hardware
//
#include <hip/hip_runtime.h>

// ---- problem dims ----
#define B_TOK 2048
#define W_DIM 512
#define NEXP  16
#define TOPK  2
#define INTER 1024
#define SHI   2048
#define OUTD  128
#define KC    32          // K-chunk per WMMA step

typedef __attribute__((ext_vector_type(16))) __bf16 v16bf;
typedef __attribute__((ext_vector_type(8)))  float  v8f;
typedef unsigned int u32;

#define AS1 __attribute__((address_space(1)))
#define AS3 __attribute__((address_space(3)))

#if defined(__gfx1250__) && __has_builtin(__builtin_amdgcn_global_load_async_to_lds_b32)
#define USE_ASYNC 1
#else
#define USE_ASYNC 0
#endif

// ---------------- helpers ----------------
static __device__ __forceinline__ u32 pk2bf(float a, float b) {
  __bf16 ha = (__bf16)a, hb = (__bf16)b;
  unsigned short ua = __builtin_bit_cast(unsigned short, ha);
  unsigned short ub = __builtin_bit_cast(unsigned short, hb);
  return (u32)ua | ((u32)ub << 16);
}

static __device__ __forceinline__ v8f zero8() {
  v8f z = {0.f, 0.f, 0.f, 0.f, 0.f, 0.f, 0.f, 0.f};
  return z;
}

// async (or fallback) 4-byte global->LDS copy
static __device__ __forceinline__ void cp_b32(void* l, const void* g) {
#if USE_ASYNC
  __builtin_amdgcn_global_load_async_to_lds_b32((AS1 int*)g, (AS3 int*)l, 0, 0);
#else
  *(u32*)l = *(const u32*)g;
#endif
}

static __device__ __forceinline__ void wait_async() {
#if USE_ASYNC
#if __has_builtin(__builtin_amdgcn_s_wait_asynccnt)
  __builtin_amdgcn_s_wait_asynccnt(0);
#else
  asm volatile("s_wait_asynccnt 0" ::: "memory");
#endif
#endif
}

// A fragment (16x32 bf16) from LDS tile (row stride 40 halves).
// lanes 0-15: row M=lane, K={0..15}; lanes 16-31: same M, K={16..31}.
static __device__ __forceinline__ v16bf frag_a(const __bf16* tile, int m, int hf) {
  v16bf a;
  u32* ap = (u32*)&a;
  const __bf16* rp = tile + m * 40 + hf * 16;
#pragma unroll
  for (int v = 0; v < 8; ++v) {
    int kb = 2 * v;
    ap[v] = *(const u32*)(rp + kb);
  }
  return a;
}

// B fragment (32x16 bf16) from LDS tile stored [n][k] (row stride 40 halves).
static __device__ __forceinline__ v16bf frag_b(const __bf16* tile, int n, int hf) {
  v16bf b;
  u32* bp = (u32*)&b;
  const __bf16* rp = tile + n * 40 + hf * 16;
#pragma unroll
  for (int v = 0; v < 8; ++v) bp[v] = *(const u32*)(rp + 2 * v);
  return b;
}

static __device__ __forceinline__ v8f wmma_bf(v16bf a, v16bf b, v8f c) {
  return __builtin_amdgcn_wmma_f32_16x16x32_bf16(false, a, false, b, (short)0, c,
                                                 false, false);
}

// Stage ROWS x 32 fp32 tile -> bf16 LDS (row stride 40). src pre-offset (row0,k0).
template <int ROWS>
static __device__ __forceinline__ void stage_f32(__bf16* lds, const float* src,
                                                 int srcStride, int tid) {
#pragma unroll
  for (int i = 0; i < ROWS / 16; ++i) {
    int idx = tid + i * 256;
    int r = idx >> 4, c = idx & 15;
    float2 f = *(const float2*)(src + (size_t)r * srcStride + 2 * c);
    ((u32*)lds)[r * 20 + c] = pk2bf(f.x, f.y);
  }
}

// Stage 64 x 32 bf16 tile (contiguous rows) -> LDS via async copies.
static __device__ __forceinline__ void stage_bf16(__bf16* lds, const __bf16* src,
                                                  int srcStride, int tid) {
#pragma unroll
  for (int i = 0; i < 4; ++i) {
    int idx = tid + i * 256;
    int r = idx >> 4, c = idx & 15;
    cp_b32((u32*)lds + r * 20 + c, (const u32*)(src + (size_t)r * srcStride) + c);
  }
}

// ---------------- kernel: zero expert counts ----------------
__global__ void k_zero(int* counts) {
  if (threadIdx.x < NEXP) counts[threadIdx.x] = 0;
}

// ---------------- kernel: gate (softmax + top2 + scatter + x->bf16) --------
__global__ __launch_bounds__(256) void k_gate(const float* __restrict__ x,
                                              const float* __restrict__ gate_w,
                                              __bf16* __restrict__ xbf,
                                              int* __restrict__ counts,
                                              int* __restrict__ toklist,
                                              float* __restrict__ wgtlist) {
  __shared__ float xs[16 * 520];
  __shared__ float lg[256];
  int tid = threadIdx.x;
  int tb = blockIdx.x * 16;
#pragma unroll
  for (int i = 0; i < 32; ++i) {
    int idx = tid + i * 256;
    int row = idx >> 9, col = idx & 511;
    float v = x[(size_t)tb * W_DIM + idx];
    xs[row * 520 + col] = v;
    xbf[(size_t)tb * W_DIM + idx] = (__bf16)v;
  }
  __syncthreads();
  int tok = tid & 15, e = tid >> 4;
  const float* gw = gate_w + (size_t)e * W_DIM;
  float acc = 0.f;
  for (int k = 0; k < W_DIM; ++k) acc += xs[tok * 520 + k] * gw[k];
  lg[e * 16 + tok] = acc;
  __syncthreads();
  if (tid < 16) {
    int t = tid;
    float mx = -1e30f;
#pragma unroll
    for (int i = 0; i < NEXP; ++i) mx = fmaxf(mx, lg[i * 16 + t]);
    float ex[NEXP], s = 0.f;
#pragma unroll
    for (int i = 0; i < NEXP; ++i) {
      ex[i] = __expf(lg[i * 16 + t] - mx);
      s += ex[i];
    }
    int i0 = 0;
    float v0 = ex[0];
#pragma unroll
    for (int i = 1; i < NEXP; ++i)
      if (ex[i] > v0) { v0 = ex[i]; i0 = i; }
    int i1 = -1;
    float v1 = -1.f;
#pragma unroll
    for (int i = 0; i < NEXP; ++i)
      if (i != i0 && ex[i] > v1) { v1 = ex[i]; i1 = i; }
    float inv = 1.f / s;  // ROUTE_SCALE == 1.0
    int g = tb + t;
    int p0 = atomicAdd(&counts[i0], 1);
    toklist[i0 * B_TOK + p0] = g;
    wgtlist[i0 * B_TOK + p0] = v0 * inv;
    int p1 = atomicAdd(&counts[i1], 1);
    toklist[i1 * B_TOK + p1] = g;
    wgtlist[i1 * B_TOK + p1] = v1 * inv;
  }
}

// ---------------- kernel: shared expert up-proj (M64 x N128, dbl-buffered) -
__global__ __launch_bounds__(256) void k_sharedA(const __bf16* __restrict__ xbf,
                                                 const float* __restrict__ sw1,
                                                 const float* __restrict__ sb1,
                                                 const float* __restrict__ sw3,
                                                 const float* __restrict__ sb3,
                                                 __bf16* __restrict__ Hs) {
  __shared__ __bf16 sA[2][64 * 40], sB1[2][128 * 40], sB3[2][128 * 40];
  int tid = threadIdx.x;
  int mbase = blockIdx.x * 64, nbase = blockIdx.y * 128;
  int lane = tid & 31, wv = tid >> 5, wm = wv & 3, wn = wv >> 2;
  int m = lane & 15, hf = lane >> 4;
  const __bf16* Asrc = xbf + (size_t)mbase * W_DIM;
  const float* B1src = sw1 + (size_t)nbase * W_DIM;
  const float* B3src = sw3 + (size_t)nbase * W_DIM;
  v8f c1[4] = {zero8(), zero8(), zero8(), zero8()};
  v8f c3[4] = {zero8(), zero8(), zero8(), zero8()};
  stage_bf16(sA[0], Asrc, W_DIM, tid);
  stage_f32<128>(sB1[0], B1src, W_DIM, tid);
  stage_f32<128>(sB3[0], B3src, W_DIM, tid);
  int p = 0;
  for (int kc = 0; kc < W_DIM; kc += KC, p ^= 1) {
    wait_async();
    __syncthreads();
    if (kc + KC < W_DIM) {
      stage_bf16(sA[p ^ 1], Asrc + kc + KC, W_DIM, tid);
      stage_f32<128>(sB1[p ^ 1], B1src + kc + KC, W_DIM, tid);
      stage_f32<128>(sB3[p ^ 1], B3src + kc + KC, W_DIM, tid);
    }
    v16bf a = frag_a(sA[p] + wm * 16 * 40, m, hf);
#pragma unroll
    for (int cn = 0; cn < 4; ++cn) {
      v16bf bf = frag_b(sB1[p] + (wn * 64 + cn * 16) * 40, m, hf);
      c1[cn] = wmma_bf(a, bf, c1[cn]);
    }
#pragma unroll
    for (int cn = 0; cn < 4; ++cn) {
      v16bf bf = frag_b(sB3[p] + (wn * 64 + cn * 16) * 40, m, hf);
      c3[cn] = wmma_bf(a, bf, c3[cn]);
    }
  }
#pragma unroll
  for (int cn = 0; cn < 4; ++cn) {
    int n = nbase + wn * 64 + cn * 16 + m;
    float bb1 = sb1[n], bb3 = sb3[n];
#pragma unroll
    for (int r = 0; r < 8; ++r) {
      int row = mbase + wm * 16 + r + hf * 8;
      float a1 = c1[cn][r] + bb1;
      float a3 = c3[cn][r] + bb3;
      float h = (a1 > 0.f ? a1 : 0.01f * a1) * a3;
      Hs[(size_t)row * SHI + n] = (__bf16)h;
    }
  }
}

// ---------------- kernel: shared expert down-proj (initializes acc) --------
__global__ __launch_bounds__(256) void k_sharedB(const __bf16* __restrict__ Hs,
                                                 const float* __restrict__ sw2,
                                                 const float* __restrict__ sb2,
                                                 float* __restrict__ acc) {
  __shared__ __bf16 sA[2][64 * 40], sB[2][128 * 40];
  int tid = threadIdx.x;
  int mbase = blockIdx.x * 64, nbase = blockIdx.y * 128;
  int lane = tid & 31, wv = tid >> 5, wm = wv & 3, wn = wv >> 2;
  int m = lane & 15, hf = lane >> 4;
  const __bf16* Asrc = Hs + (size_t)mbase * SHI;
  const float* Bsrc = sw2 + (size_t)nbase * SHI;
  v8f c[4] = {zero8(), zero8(), zero8(), zero8()};
  stage_bf16(sA[0], Asrc, SHI, tid);
  stage_f32<128>(sB[0], Bsrc, SHI, tid);
  int p = 0;
  for (int kc = 0; kc < SHI; kc += KC, p ^= 1) {
    wait_async();
    __syncthreads();
    if (kc + KC < SHI) {
      stage_bf16(sA[p ^ 1], Asrc + kc + KC, SHI, tid);
      stage_f32<128>(sB[p ^ 1], Bsrc + kc + KC, SHI, tid);
    }
    v16bf a = frag_a(sA[p] + wm * 16 * 40, m, hf);
#pragma unroll
    for (int cn = 0; cn < 4; ++cn) {
      v16bf bf = frag_b(sB[p] + (wn * 64 + cn * 16) * 40, m, hf);
      c[cn] = wmma_bf(a, bf, c[cn]);
    }
  }
#pragma unroll
  for (int cn = 0; cn < 4; ++cn) {
    int n = nbase + wn * 64 + cn * 16 + m;
    float bb = sb2[n];
#pragma unroll
    for (int r = 0; r < 8; ++r) {
      int row = mbase + wm * 16 + r + hf * 8;
      acc[(size_t)row * W_DIM + n] = c[cn][r] + bb;
    }
  }
}

// ---------------- kernel: routed experts up-proj (sparse, gathered) --------
__global__ __launch_bounds__(256) void k_routedA(const __bf16* __restrict__ xbf,
                                                 const float* __restrict__ w1,
                                                 const float* __restrict__ b1,
                                                 const float* __restrict__ w3,
                                                 const float* __restrict__ b3,
                                                 const int* __restrict__ counts,
                                                 const int* __restrict__ toklist,
                                                 __bf16* __restrict__ Hr) {
  __shared__ __bf16 sA[2][64 * 40], sB1[2][128 * 40], sB3[2][128 * 40];
  __shared__ int sTok[64];
  __shared__ int sOff, sCnt;
  int tid = threadIdx.x;
  int e = blockIdx.x, nbase = blockIdx.y * 128;
  if (tid == 0) {
    int o = 0;
    for (int i = 0; i < e; ++i) o += counts[i];
    sOff = o;
    sCnt = counts[e];
  }
  __syncthreads();
  int n_e = sCnt, off = sOff;
  int lane = tid & 31, wv = tid >> 5, wm = wv & 3, wn = wv >> 2;
  int m = lane & 15, hf = lane >> 4;
  const float* w1e = w1 + ((size_t)e * INTER + nbase) * W_DIM;
  const float* w3e = w3 + ((size_t)e * INTER + nbase) * W_DIM;
  for (int mt = 0; mt * 64 < n_e; ++mt) {
    __syncthreads();
    if (tid < 64) {
      int slot = mt * 64 + tid;
      if (slot >= n_e) slot = n_e - 1;
      sTok[tid] = toklist[e * B_TOK + slot];
    }
    __syncthreads();
    v8f c1[4] = {zero8(), zero8(), zero8(), zero8()};
    v8f c3[4] = {zero8(), zero8(), zero8(), zero8()};
    // prologue stage chunk 0
#pragma unroll
    for (int i = 0; i < 4; ++i) {
      int idx = tid + i * 256;
      int r = idx >> 4, cidx = idx & 15;
      cp_b32((u32*)sA[0] + r * 20 + cidx,
             (const u32*)(xbf + (size_t)sTok[r] * W_DIM) + cidx);
    }
    stage_f32<128>(sB1[0], w1e, W_DIM, tid);
    stage_f32<128>(sB3[0], w3e, W_DIM, tid);
    int p = 0;
    for (int kc = 0; kc < W_DIM; kc += KC, p ^= 1) {
      wait_async();
      __syncthreads();
      if (kc + KC < W_DIM) {
#pragma unroll
        for (int i = 0; i < 4; ++i) {
          int idx = tid + i * 256;
          int r = idx >> 4, cidx = idx & 15;
          cp_b32((u32*)sA[p ^ 1] + r * 20 + cidx,
                 (const u32*)(xbf + (size_t)sTok[r] * W_DIM + kc + KC) + cidx);
        }
        stage_f32<128>(sB1[p ^ 1], w1e + kc + KC, W_DIM, tid);
        stage_f32<128>(sB3[p ^ 1], w3e + kc + KC, W_DIM, tid);
      }
      v16bf a = frag_a(sA[p] + wm * 16 * 40, m, hf);
#pragma unroll
      for (int cn = 0; cn < 4; ++cn) {
        v16bf bf = frag_b(sB1[p] + (wn * 64 + cn * 16) * 40, m, hf);
        c1[cn] = wmma_bf(a, bf, c1[cn]);
      }
#pragma unroll
      for (int cn = 0; cn < 4; ++cn) {
        v16bf bf = frag_b(sB3[p] + (wn * 64 + cn * 16) * 40, m, hf);
        c3[cn] = wmma_bf(a, bf, c3[cn]);
      }
    }
#pragma unroll
    for (int cn = 0; cn < 4; ++cn) {
      int n = nbase + wn * 64 + cn * 16 + m;
      float bb1 = b1[(size_t)e * INTER + n], bb3 = b3[(size_t)e * INTER + n];
#pragma unroll
      for (int r = 0; r < 8; ++r) {
        int slot = mt * 64 + wm * 16 + r + hf * 8;
        if (slot < n_e) {
          float a1 = c1[cn][r] + bb1;
          float a3 = c3[cn][r] + bb3;
          float h = (a1 > 0.f ? a1 : 0.01f * a1) * a3;
          Hr[(size_t)(off + slot) * INTER + n] = (__bf16)h;
        }
      }
    }
  }
}

// ---------------- kernel: routed experts down-proj + weighted combine ------
__global__ __launch_bounds__(256) void k_routedB(const __bf16* __restrict__ Hr,
                                                 const float* __restrict__ w2,
                                                 const float* __restrict__ b2,
                                                 const int* __restrict__ counts,
                                                 const int* __restrict__ toklist,
                                                 const float* __restrict__ wgtlist,
                                                 float* __restrict__ acc) {
  __shared__ __bf16 sA[2][64 * 40], sB[2][128 * 40];
  __shared__ int sTok[64];
  __shared__ float sWgt[64];
  __shared__ int sOff, sCnt;
  int tid = threadIdx.x;
  int e = blockIdx.x, nbase = blockIdx.y * 128;
  if (tid == 0) {
    int o = 0;
    for (int i = 0; i < e; ++i) o += counts[i];
    sOff = o;
    sCnt = counts[e];
  }
  __syncthreads();
  int n_e = sCnt, off = sOff;
  int lane = tid & 31, wv = tid >> 5, wm = wv & 3, wn = wv >> 2;
  int m = lane & 15, hf = lane >> 4;
  const float* w2e = w2 + ((size_t)e * W_DIM + nbase) * INTER;
  for (int mt = 0; mt * 64 < n_e; ++mt) {
    __syncthreads();
    if (tid < 64) {
      int slot = mt * 64 + tid;
      if (slot >= n_e) slot = n_e - 1;
      sTok[tid] = toklist[e * B_TOK + slot];
      sWgt[tid] = wgtlist[e * B_TOK + slot];
    }
    __syncthreads();
    v8f c[4] = {zero8(), zero8(), zero8(), zero8()};
    const __bf16* Abase = Hr + (size_t)off * INTER;
#pragma unroll
    for (int i = 0; i < 4; ++i) {
      int idx = tid + i * 256;
      int r = idx >> 4, cidx = idx & 15;
      int slot = mt * 64 + r;
      if (slot >= n_e) slot = n_e - 1;
      cp_b32((u32*)sA[0] + r * 20 + cidx,
             (const u32*)(Abase + (size_t)slot * INTER) + cidx);
    }
    stage_f32<128>(sB[0], w2e, INTER, tid);
    int p = 0;
    for (int kc = 0; kc < INTER; kc += KC, p ^= 1) {
      wait_async();
      __syncthreads();
      if (kc + KC < INTER) {
#pragma unroll
        for (int i = 0; i < 4; ++i) {
          int idx = tid + i * 256;
          int r = idx >> 4, cidx = idx & 15;
          int slot = mt * 64 + r;
          if (slot >= n_e) slot = n_e - 1;
          cp_b32((u32*)sA[p ^ 1] + r * 20 + cidx,
                 (const u32*)(Abase + (size_t)slot * INTER + kc + KC) + cidx);
        }
        stage_f32<128>(sB[p ^ 1], w2e + kc + KC, INTER, tid);
      }
      v16bf a = frag_a(sA[p] + wm * 16 * 40, m, hf);
#pragma unroll
      for (int cn = 0; cn < 4; ++cn) {
        v16bf bf = frag_b(sB[p] + (wn * 64 + cn * 16) * 40, m, hf);
        c[cn] = wmma_bf(a, bf, c[cn]);
      }
    }
#pragma unroll
    for (int cn = 0; cn < 4; ++cn) {
      int n = nbase + wn * 64 + cn * 16 + m;
      float bb = b2[(size_t)e * W_DIM + n];
#pragma unroll
      for (int r = 0; r < 8; ++r) {
        int ti = wm * 16 + r + hf * 8;
        int slot = mt * 64 + ti;
        if (slot < n_e) {
          float eo = c[cn][r] + bb;
          atomicAdd(acc + (size_t)sTok[ti] * W_DIM + n, sWgt[ti] * eo);
        }
      }
    }
  }
}

// ---------------- kernel: output projection (N = 128 exactly) --------------
__global__ __launch_bounds__(256) void k_out(const float* __restrict__ acc,
                                             const float* __restrict__ ow,
                                             const float* __restrict__ ob,
                                             float* __restrict__ out) {
  __shared__ __bf16 sA[2][64 * 40], sB[2][128 * 40];
  int tid = threadIdx.x;
  int mbase = blockIdx.x * 64;
  int lane = tid & 31, wv = tid >> 5, wm = wv & 3, wn = wv >> 2;
  int m = lane & 15, hf = lane >> 4;
  const float* Asrc = acc + (size_t)mbase * W_DIM;
  v8f c[4] = {zero8(), zero8(), zero8(), zero8()};
  stage_f32<64>(sA[0], Asrc, W_DIM, tid);
  stage_f32<128>(sB[0], ow, W_DIM, tid);
  int p = 0;
  for (int kc = 0; kc < W_DIM; kc += KC, p ^= 1) {
    __syncthreads();
    if (kc + KC < W_DIM) {
      stage_f32<64>(sA[p ^ 1], Asrc + kc + KC, W_DIM, tid);
      stage_f32<128>(sB[p ^ 1], ow + kc + KC, W_DIM, tid);
    }
    v16bf a = frag_a(sA[p] + wm * 16 * 40, m, hf);
#pragma unroll
    for (int cn = 0; cn < 4; ++cn) {
      v16bf bf = frag_b(sB[p] + (wn * 64 + cn * 16) * 40, m, hf);
      c[cn] = wmma_bf(a, bf, c[cn]);
    }
  }
#pragma unroll
  for (int cn = 0; cn < 4; ++cn) {
    int n = wn * 64 + cn * 16 + m;
    float bb = ob[n];
#pragma unroll
    for (int r = 0; r < 8; ++r) {
      int row = mbase + wm * 16 + r + hf * 8;
      out[(size_t)row * OUTD + n] = c[cn][r] + bb;
    }
  }
}

// ---------------- host launcher ----------------
extern "C" void kernel_launch(void* const* d_in, const int* in_sizes, int n_in,
                              void* d_out, int out_size, void* d_ws, size_t ws_size,
                              hipStream_t stream) {
  const float* x      = (const float*)d_in[0];
  // d_in[1] = task_id (unused)
  const float* gate_w = (const float*)d_in[2];
  const float* w1     = (const float*)d_in[3];
  const float* b1     = (const float*)d_in[4];
  const float* w2     = (const float*)d_in[5];
  const float* b2     = (const float*)d_in[6];
  const float* w3     = (const float*)d_in[7];
  const float* b3     = (const float*)d_in[8];
  const float* sw1    = (const float*)d_in[9];
  const float* sb1    = (const float*)d_in[10];
  const float* sw2    = (const float*)d_in[11];
  const float* sb2    = (const float*)d_in[12];
  const float* sw3    = (const float*)d_in[13];
  const float* sb3    = (const float*)d_in[14];
  const float* ow     = (const float*)d_in[15];
  const float* ob     = (const float*)d_in[16];
  float* out = (float*)d_out;

  char* w = (char*)d_ws;
  __bf16* xbf = (__bf16*)w;   w += (size_t)B_TOK * W_DIM * 2;
  __bf16* Hr  = (__bf16*)w;   w += (size_t)B_TOK * TOPK * INTER * 2;
  __bf16* Hs  = (__bf16*)w;   w += (size_t)B_TOK * SHI * 2;
  float*  acc = (float*)w;    w += (size_t)B_TOK * W_DIM * 4;
  int*   counts  = (int*)w;   w += 256;
  int*   toklist = (int*)w;   w += (size_t)NEXP * B_TOK * 4;
  float* wgtlist = (float*)w; w += (size_t)NEXP * B_TOK * 4;

  dim3 blk(256);
  k_zero<<<1, 32, 0, stream>>>(counts);
  k_gate<<<B_TOK / 16, blk, 0, stream>>>(x, gate_w, xbf, counts, toklist, wgtlist);
  k_sharedA<<<dim3(B_TOK / 64, SHI / 128), blk, 0, stream>>>(xbf, sw1, sb1, sw3, sb3, Hs);
  k_sharedB<<<dim3(B_TOK / 64, W_DIM / 128), blk, 0, stream>>>(Hs, sw2, sb2, acc);
  k_routedA<<<dim3(NEXP, INTER / 128), blk, 0, stream>>>(xbf, w1, b1, w3, b3, counts,
                                                         toklist, Hr);
  k_routedB<<<dim3(NEXP, W_DIM / 128), blk, 0, stream>>>(Hr, w2, b2, counts, toklist,
                                                         wgtlist, acc);
  k_out<<<dim3(B_TOK / 64, 1), blk, 0, stream>>>(acc, ow, ob, out);
}
